// MultiHeadAttention_12386685681677
// MI455X (gfx1250) — matrix operand order repaired and checked by reading, not compiled, not run on hardware
//
#include <hip/hip_runtime.h>

// ---------------------------------------------------------------------------
// AlphaFold-style pair-biased MHA for MI455X (gfx1250, wave32, WMMA).
// B=4, L=1024, node=1024, H=16, D=64, pair=64.
// HBM roofline: z (1.07 GB) dominates -> read exactly once, never materialize
// the [B,H,L,L] bias. Projection GEMMs: LDS-blocked 64x128 macro-tiles with
// double-buffered async global->LDS staging (global_load_async_to_lds_b128 +
// s_wait_asynccnt). Attention: bf16 WMMA; pair-bias via v_wmma_f32_16x16x4_f32
// streamed straight from f32 z in global memory.
// ---------------------------------------------------------------------------

#define BDIM 4
#define LSEQ 1024
#define ND   1024
#define NH   16
#define HD   64
#define PD   64

// GEMM macro-tile config
#define GM   64
#define GN   128
#define GK   64
#define GKP  72   // padded LDS row stride (halves): 144B rows -> conflict-free

typedef __attribute__((ext_vector_type(16))) __bf16 v16bf;
typedef __attribute__((ext_vector_type(8)))  float  v8f;
typedef __attribute__((ext_vector_type(2)))  float  v2f;
typedef unsigned int       u32;
typedef unsigned long long u64;

static __device__ __forceinline__ __bf16 f2bf(float f) {
  // round-to-nearest-even f32 -> bf16
  u32 u = __builtin_bit_cast(u32, f);
  u32 r = (u + 0x7fffu + ((u >> 16) & 1u)) >> 16;
  unsigned short s = (unsigned short)r;
  return __builtin_bit_cast(__bf16, s);
}

static __device__ __forceinline__ v8f wmma_bf16(v16bf a, v16bf b, v8f c) {
  return __builtin_amdgcn_wmma_f32_16x16x32_bf16(false, a, false, b, (short)0, c,
                                                 false, false);
}
static __device__ __forceinline__ v8f wmma_f32(v2f a, v2f b, v8f c) {
  return __builtin_amdgcn_wmma_f32_16x16x4_f32(false, a, false, b, (short)0, c,
                                               false, false);
}

// Layout within each 32-column block so a wave32 bf16 A-fragment (16x32) is
// one contiguous 32B run per lane:
//   half0 lanes read K={0..15}  at element offsets  0..15
//   half1 lanes read K={16..31} at element offsets 16..31
static __device__ __forceinline__ int perm32(int c) {
  return c;
}
static __device__ __forceinline__ int swz(int c) {
  return (c & ~31) | perm32(c & 31);
}

// ---------------------------------------------------------------------------
// f32 -> bf16 converts
// ---------------------------------------------------------------------------
__global__ void cvt_plain_kernel(const float* __restrict__ src,
                                 __bf16* __restrict__ dst, int n) {
  int stride = gridDim.x * blockDim.x;
  for (int i = blockIdx.x * blockDim.x + threadIdx.x; i < n; i += stride)
    dst[i] = f2bf(src[i]);
}

// rows x 1024, columns swizzled for A-fragment loads
__global__ void cvt_swizzle_kernel(const float* __restrict__ src,
                                   __bf16* __restrict__ dst, int n) {
  int stride = gridDim.x * blockDim.x;
  for (int i = blockIdx.x * blockDim.x + threadIdx.x; i < n; i += stride) {
    int r = i >> 10, c = i & 1023;
    dst[(r << 10) | swz(c)] = f2bf(src[i]);
  }
}

// ---------------------------------------------------------------------------
// LDS-blocked GEMM core (8 waves, macro 64x128, wave 32x32, K-stage 64).
// A: [M][K] bf16 pre-swizzled; Bw: [N][K] bf16 plain row-major.
// Staging uses CDNA5 async global->LDS (ASYNCcnt), double-buffered.
// ---------------------------------------------------------------------------
static __device__ __forceinline__ void gemm_stage_async(
    const __bf16* Ab, const __bf16* Bb, __bf16* As, __bf16* Bs, int K,
    int kst, int tid) {
  // A: 64 rows x 128B = 512 x 16B chunks, 2 per thread
#pragma unroll
  for (int i = 0; i < 2; ++i) {
    int c = tid + 256 * i;
    int row = c >> 3, seg = c & 7;
    u32 goff = (u32)((row * K + kst + seg * 8) * 2);
    u32 lds  = (u32)(u64)(As + row * GKP + seg * 8);
    asm volatile("global_load_async_to_lds_b128 %0, %1, %2"
                 :: "v"(lds), "v"(goff), "s"((u64)Ab) : "memory");
  }
  // B: 128 rows x 128B = 1024 x 16B chunks, 4 per thread
#pragma unroll
  for (int i = 0; i < 4; ++i) {
    int c = tid + 256 * i;
    int row = c >> 3, seg = c & 7;
    u32 goff = (u32)((row * K + kst + seg * 8) * 2);
    u32 lds  = (u32)(u64)(Bs + row * GKP + seg * 8);
    asm volatile("global_load_async_to_lds_b128 %0, %1, %2"
                 :: "v"(lds), "v"(goff), "s"((u64)Bb) : "memory");
  }
}

static __device__ __forceinline__ void gemm_compute_stage(
    const __bf16* As, const __bf16* Bs, int wy, int wx, int ln, int half,
    v8f acc[2][2]) {
#pragma unroll
  for (int kk = 0; kk < 2; ++kk) {
    v16bf a0 = *(const v16bf*)(As + (wy * 32 + ln) * GKP + kk * 32 + 16 * half);
    v16bf a1 = *(const v16bf*)(As + (wy * 32 + 16 + ln) * GKP + kk * 32 + 16 * half);
    v16bf b0 = *(const v16bf*)(Bs + (wx * 32 + ln) * GKP + kk * 32 + 16 * half);
    v16bf b1 = *(const v16bf*)(Bs + (wx * 32 + 16 + ln) * GKP + kk * 32 + 16 * half);
    acc[0][0] = wmma_bf16(a0, b0, acc[0][0]);
    acc[0][1] = wmma_bf16(a0, b1, acc[0][1]);
    acc[1][0] = wmma_bf16(a1, b0, acc[1][0]);
    acc[1][1] = wmma_bf16(a1, b1, acc[1][1]);
  }
}

// ---------------------------------------------------------------------------
// QKV projection: [4096 x 1024] @ [3072 x 1024]^T
// q -> q_ws [B][H][L][D]  (D swizzled: A-side of QK^T)
// k -> k_ws [B][H][L][D]  (plain: B-side of QK^T)
// v -> vT_ws [B][H][D][L] (transposed: B-side of PV)
// ---------------------------------------------------------------------------
__global__ __launch_bounds__(256) void qkv_gemm_kernel(
    const __bf16* __restrict__ s_sw, const __bf16* __restrict__ Wqkv,
    __bf16* __restrict__ q_ws, __bf16* __restrict__ k_ws,
    __bf16* __restrict__ vT_ws) {
  __shared__ __bf16 As[2][GM * GKP];
  __shared__ __bf16 Bs[2][GN * GKP];
  const int K = ND;
  int tid = threadIdx.x;
  int w = tid >> 5, lane = tid & 31, ln = lane & 15, half = lane >> 4;
  int wy = w >> 2, wx = w & 3;
  const int NB = (3 * ND) / GN;  // 24 macro-cols
  int bm = (blockIdx.x / NB) * GM;
  int bn = (blockIdx.x % NB) * GN;
  const __bf16* Ab = s_sw + (u64)bm * K;
  const __bf16* Bb = Wqkv + (u64)bn * K;

  v8f acc[2][2] = {};
  gemm_stage_async(Ab, Bb, As[0], Bs[0], K, 0, tid);
  const int NK = K / GK;  // 16 stages
  for (int ks = 0; ks < NK; ++ks) {
    int cur = ks & 1;
    asm volatile("s_wait_asynccnt 0x0" ::: "memory");
    __syncthreads();  // buf[cur] populated; everyone done reading buf[cur^1]
    if (ks + 1 < NK)
      gemm_stage_async(Ab, Bb, As[cur ^ 1], Bs[cur ^ 1], K, (ks + 1) * GK, tid);
    gemm_compute_stage(As[cur], Bs[cur], wy, wx, ln, half, acc);
  }

#pragma unroll
  for (int sm = 0; sm < 2; ++sm)
#pragma unroll
    for (int sn = 0; sn < 2; ++sn) {
      int m0 = bm + wy * 32 + sm * 16;
      int n0 = bn + wx * 32 + sn * 16;
      int col = n0 + ln;
      int seg = col >> 10;  // 0=q 1=k 2=v
      int cc = col & 1023;
      int h = cc >> 6, d = cc & 63;
#pragma unroll
      for (int r = 0; r < 8; ++r) {
        int row = m0 + r + 8 * half;
        int b = row >> 10, l = row & 1023;
        __bf16 v = f2bf(acc[sm][sn][r]);
        if (seg == 0)
          q_ws[((u64)(b * NH + h) * LSEQ + l) * HD + swz(d)] = v;
        else if (seg == 1)
          k_ws[((u64)(b * NH + h) * LSEQ + l) * HD + d] = v;
        else
          vT_ws[((u64)(b * NH + h) * HD + d) * LSEQ + l] = v;
      }
    }
}

// ---------------------------------------------------------------------------
// Fused pair-bias + flash attention.
// grid = B * L/16 workgroups; block = 512 threads = 16 waves (wave == head).
// Per 32-key step:
//   * cooperative bias tile [512 pair-rows x 16 heads] via
//     v_wmma_f32_16x16x4_f32 with A streamed straight from f32 z in global
//     (4 KB active footprint/wave -> one HBM fetch per line, rest cache hits);
//   * per-head QK^T (bf16 WMMA), online softmax (shfl reductions),
//     P re-layout through wave-local LDS, PV (bf16 WMMA, full 16x32 K).
// z is read exactly once; next key-block prefetched (global_prefetch_b8).
// ---------------------------------------------------------------------------
__global__ __launch_bounds__(512) void attn_kernel(
    const float* __restrict__ z, const float* __restrict__ Wbias,
    const float* __restrict__ b_bias, const __bf16* __restrict__ q_ws,
    const __bf16* __restrict__ k_ws, const __bf16* __restrict__ vT_ws,
    __bf16* __restrict__ attn_sw) {
  __shared__ float  biasT[512 * 16];   // 32 KB  [pair_row][head]
  __shared__ __bf16 pscr[16 * 512];    // 16 KB  per-wave 16x32 P scratch

  int b  = blockIdx.x >> 6;
  int q0 = (blockIdx.x & 63) << 4;
  int tid = threadIdx.x;
  int w = tid >> 5;                    // wave id == head
  int lane = tid & 31, ln = lane & 15, half = lane >> 4;

  // W_bias B-fragments for f32 WMMA (4x16 tile each, K=64 total), in registers.
  v2f wbf[16];
#pragma unroll
  for (int kk = 0; kk < 16; ++kk)
    wbf[kk] = *(const v2f*)(Wbias + ln * PD + kk * 4 + 2 * half);

  const __bf16* qrow = q_ws + ((u64)(b * NH + w) * LSEQ + q0 + ln) * HD;
  v16bf qa0 = *(const v16bf*)(qrow + 16 * half);
  v16bf qa1 = *(const v16bf*)(qrow + 32 + 16 * half);
  float bb = b_bias[w];

  float m[8], l[8];
  v8f o[4];
#pragma unroll
  for (int r = 0; r < 8; ++r) { m[r] = -3.0e38f; l[r] = 0.0f; }
#pragma unroll
  for (int t = 0; t < 4; ++t) o[t] = v8f{};

  const float scale = 0.125f;  // 1/sqrt(64)
  const float* zq = z + (u64)(b * LSEQ + q0) * LSEQ * PD;

  for (int kb = 0; kb < LSEQ / 32; ++kb) {
    int k0 = kb << 5;
    __syncthreads();  // biasT consumers of previous step are done

    if (kb + 1 < LSEQ / 32) {
      const float* pf =
          zq + ((u64)(tid >> 5) * LSEQ + (k0 + 32) + (tid & 31)) * PD;
      __builtin_prefetch(pf, 0, 3);
      __builtin_prefetch(pf + 32, 0, 3);
    }

    // ---- cooperative bias: wave w -> pair rows 32w..32w+31, all 16 heads ----
#pragma unroll
    for (int t = 0; t < 2; ++t) {
      int m0 = w * 32 + t * 16;
      int pr = m0 + ln;                 // pair row = qi*32 + ki
      int qi = pr >> 5, ki = pr & 31;
      const float* zr = zq + ((u64)qi * LSEQ + k0 + ki) * PD + 2 * half;
      v8f bt = {};
#pragma unroll
      for (int kk = 0; kk < 16; ++kk)
        bt = wmma_f32(*(const v2f*)(zr + kk * 4), wbf[kk], bt);
#pragma unroll
      for (int r = 0; r < 8; ++r)
        biasT[(m0 + r + 8 * half) * 16 + ln] = bt[r];
    }
    __syncthreads();

    // ---- per-head scores: Q (16x64) @ K^T (64x32) -> two 16x16 C tiles ----
    const __bf16* kbase = k_ws + ((u64)(b * NH + w) * LSEQ + k0) * HD;
    const __bf16* kr0 = kbase + (u64)ln * HD + 16 * half;
    const __bf16* kr1 = kbase + (u64)(16 + ln) * HD + 16 * half;
    v8f sc0 = {}, sc1 = {};
    sc0 = wmma_bf16(qa0, *(const v16bf*)(kr0), sc0);
    sc0 = wmma_bf16(qa1, *(const v16bf*)(kr0 + 32), sc0);
    sc1 = wmma_bf16(qa0, *(const v16bf*)(kr1), sc1);
    sc1 = wmma_bf16(qa1, *(const v16bf*)(kr1 + 32), sc1);

    // ---- online softmax over 32 keys ----
    float alpha[8], p0[8], p1[8];
#pragma unroll
    for (int r = 0; r < 8; ++r) {
      int prb = (r + 8 * half) * 32;
      float x0 = sc0[r] * scale + biasT[(prb + ln) * 16 + w] + bb;
      float x1 = sc1[r] * scale + biasT[(prb + 16 + ln) * 16 + w] + bb;
      float mx = fmaxf(x0, x1);
      mx = fmaxf(mx, __shfl_xor(mx, 1, 16));
      mx = fmaxf(mx, __shfl_xor(mx, 2, 16));
      mx = fmaxf(mx, __shfl_xor(mx, 4, 16));
      mx = fmaxf(mx, __shfl_xor(mx, 8, 16));
      float mn = fmaxf(m[r], mx);
      float al = __expf(m[r] - mn);
      float e0 = __expf(x0 - mn);
      float e1 = __expf(x1 - mn);
      float rs = e0 + e1;
      rs += __shfl_xor(rs, 1, 16);
      rs += __shfl_xor(rs, 2, 16);
      rs += __shfl_xor(rs, 4, 16);
      rs += __shfl_xor(rs, 8, 16);
      l[r] = l[r] * al + rs;
      m[r] = mn;
      alpha[r] = al;
      p0[r] = e0;
      p1[r] = e1;
    }
#pragma unroll
    for (int t = 0; t < 4; ++t)
#pragma unroll
      for (int r = 0; r < 8; ++r) o[t][r] *= alpha[r];

    // ---- re-layout P (C layout -> A-fragment) through wave-local LDS ----
    __bf16* pw = pscr + w * 512;
#pragma unroll
    for (int r = 0; r < 8; ++r) {
      int qi = r + 8 * half;
      pw[qi * 32 + perm32(ln)]      = f2bf(p0[r]);
      pw[qi * 32 + perm32(16 + ln)] = f2bf(p1[r]);
    }
    asm volatile("" ::: "memory");  // keep DS store->load order (same wave)
    v16bf pa = *(const v16bf*)(pw + ln * 32 + 16 * half);

    // ---- PV: P (16x32) @ V (32x16 per d-tile) ----
    const __bf16* vb =
        vT_ws + ((u64)(b * NH + w) * HD + ln) * LSEQ + k0 + 16 * half;
#pragma unroll
    for (int t = 0; t < 4; ++t)
      o[t] = wmma_bf16(pa, *(const v16bf*)(vb + (u64)(t * 16) * LSEQ), o[t]);
  }

  // ---- epilogue: normalize, store swizzled bf16 rows for out-proj A-side ----
#pragma unroll
  for (int t = 0; t < 4; ++t)
#pragma unroll
    for (int r = 0; r < 8; ++r) {
      int qi = r + 8 * half;
      int c = w * HD + t * 16 + ln;
      attn_sw[(u64)(b * LSEQ + q0 + qi) * ND + swz(c)] = f2bf(o[t][r] / l[r]);
    }
}

// ---------------------------------------------------------------------------
// Output projection: [4096 x 1024] @ [1024 x 1024]^T + b_out -> f32 out
// ---------------------------------------------------------------------------
__global__ __launch_bounds__(256) void outproj_kernel(
    const __bf16* __restrict__ attn_sw, const __bf16* __restrict__ Wout,
    const float* __restrict__ b_out, float* __restrict__ out) {
  __shared__ __bf16 As[2][GM * GKP];
  __shared__ __bf16 Bs[2][GN * GKP];
  const int K = ND;
  int tid = threadIdx.x;
  int w = tid >> 5, lane = tid & 31, ln = lane & 15, half = lane >> 4;
  int wy = w >> 2, wx = w & 3;
  const int NB = ND / GN;  // 8 macro-cols
  int bm = (blockIdx.x / NB) * GM;
  int bn = (blockIdx.x % NB) * GN;
  const __bf16* Ab = attn_sw + (u64)bm * K;
  const __bf16* Bb = Wout + (u64)bn * K;

  v8f acc[2][2] = {};
  gemm_stage_async(Ab, Bb, As[0], Bs[0], K, 0, tid);
  const int NK = K / GK;
  for (int ks = 0; ks < NK; ++ks) {
    int cur = ks & 1;
    asm volatile("s_wait_asynccnt 0x0" ::: "memory");
    __syncthreads();
    if (ks + 1 < NK)
      gemm_stage_async(Ab, Bb, As[cur ^ 1], Bs[cur ^ 1], K, (ks + 1) * GK, tid);
    gemm_compute_stage(As[cur], Bs[cur], wy, wx, ln, half, acc);
  }

#pragma unroll
  for (int sm = 0; sm < 2; ++sm)
#pragma unroll
    for (int sn = 0; sn < 2; ++sn) {
      int m0 = bm + wy * 32 + sm * 16;
      int n0 = bn + wx * 32 + sn * 16;
      int col = n0 + ln;
      float bias = b_out[col];
#pragma unroll
      for (int r = 0; r < 8; ++r) {
        int row = m0 + r + 8 * half;
        out[(u64)row * ND + col] = acc[sm][sn][r] + bias;
      }
    }
}

// ---------------------------------------------------------------------------
// Launch
// ---------------------------------------------------------------------------
extern "C" void kernel_launch(void* const* d_in, const int* in_sizes, int n_in,
                              void* d_out, int out_size, void* d_ws,
                              size_t ws_size, hipStream_t stream) {
  const float* s      = (const float*)d_in[0];
  const float* z      = (const float*)d_in[1];
  const float* W_qkv  = (const float*)d_in[2];
  const float* W_bias = (const float*)d_in[3];
  const float* b_bias = (const float*)d_in[4];
  const float* W_out  = (const float*)d_in[5];
  const float* b_out  = (const float*)d_in[6];
  float* out = (float*)d_out;

  char* ws = (char*)d_ws;
  const size_t SZ_S    = (size_t)BDIM * LSEQ * ND * 2;       // 8 MB
  const size_t SZ_WQKV = (size_t)3 * ND * ND * 2;            // 6 MB
  const size_t SZ_WOUT = (size_t)ND * ND * 2;                // 2 MB
  const size_t SZ_HLD  = (size_t)BDIM * NH * LSEQ * HD * 2;  // 8 MB
  size_t off = 0;
  __bf16* s_sw    = (__bf16*)(ws + off); off += SZ_S;
  __bf16* wqkv    = (__bf16*)(ws + off); off += SZ_WQKV;
  __bf16* wout    = (__bf16*)(ws + off); off += SZ_WOUT;
  __bf16* q_ws    = (__bf16*)(ws + off); off += SZ_HLD;
  __bf16* k_ws    = (__bf16*)(ws + off); off += SZ_HLD;
  __bf16* vT_ws   = (__bf16*)(ws + off); off += SZ_HLD;
  __bf16* attn_sw = (__bf16*)(ws + off); off += SZ_S;
  (void)ws_size; (void)in_sizes; (void)n_in; (void)out_size;

  cvt_swizzle_kernel<<<1024, 256, 0, stream>>>(s, s_sw, BDIM * LSEQ * ND);
  cvt_plain_kernel<<<1024, 256, 0, stream>>>(W_qkv, wqkv, 3 * ND * ND);
  cvt_plain_kernel<<<1024, 256, 0, stream>>>(W_out, wout, ND * ND);

  // QKV GEMM: 64 x 24 macro-tiles (64x128), async double-buffered LDS
  qkv_gemm_kernel<<<64 * 24, 256, 0, stream>>>(s_sw, wqkv, q_ws, k_ws, vT_ws);

  // fused bias + attention: one WG per (b, 16-query tile), 16 waves = heads
  attn_kernel<<<BDIM * (LSEQ / 16), 512, 0, stream>>>(z, W_bias, b_bias, q_ws,
                                                      k_ws, vT_ws, attn_sw);

  // output projection: 64 x 8 macro-tiles
  outproj_kernel<<<64 * 8, 256, 0, stream>>>(attn_sw, wout, b_out, out);
}
